// MultiHeadedAttention_28827820491301
// MI455X (gfx1250) — matrix operand order repaired and checked by reading, not compiled, not run on hardware
//
#include <hip/hip_runtime.h>

// ---------------- problem constants ----------------
#define BB   4
#define SS   2048
#define DD   1024
#define HH   16
#define DKK  64
#define NEGV (-1000000000.0f)

// ---------------- WMMA types ----------------
typedef __bf16 v16bf __attribute__((ext_vector_type(16)));
typedef __bf16 v8bf  __attribute__((ext_vector_type(8)));
typedef float  v8f   __attribute__((ext_vector_type(8)));

union Frag16 { v16bf v; unsigned int u[8]; };

__device__ __forceinline__ v8f wmma_bf16(v16bf a, v16bf b, v8f c) {
  // D(16x16,f32) = A(16x32,bf16) * B(32x16,bf16) + C
  return __builtin_amdgcn_wmma_f32_16x16x32_bf16(
      /*neg_a=*/false, a, /*neg_b=*/false, b,
      /*c_mod=*/(short)0, c, /*reuse_a=*/false, /*reuse_b=*/false);
}

// A-fragment (16-bit, 16x32) K index of element pair j for half h:
//   VGPR0..3: K = 2j + 8h ; VGPR4..7: K = 16 + 2(j-4) + 8h
// (j=0..3 and j=4..7 are each 4 consecutive dwords -> two b128 loads)
__device__ __forceinline__ constexpr int kbA(int j, int h) {
  return 2 * (j & 3) + (h << 3) + ((j >> 2) << 4);
}
// B-fragment (16-bit, 32x16) K index: same K order as the A-fragment (kbA).
__device__ __forceinline__ constexpr int kbB(int j, int h) {
  return 2 * (j & 3) + (h << 3) + ((j >> 2) << 4);
}

// ---------------- fp32 -> bf16 convert ----------------
__global__ __launch_bounds__(256) void cvt_f32_to_bf16(
    const float* __restrict__ src, __bf16* __restrict__ dst, int n) {
  int i = blockIdx.x * blockDim.x + threadIdx.x;
  const int stride = gridDim.x * blockDim.x;
  for (; i < n; i += stride) dst[i] = (__bf16)src[i];
}

// ---------------- NT GEMM: Y = A(MxK) * W^T + bias ----------------
// A row-major, W row-major (rows of W == columns of W^T) -> pure NT form.
// One wave computes a 16x64 output tile. M=BB*SS, N=K=DD fixed.
// ABF16: 0 = A is fp32 (converted in-register), 1 = A is bf16.
// MODE : 0 = store bf16 [M][N]
//        1 = store bf16 transposed per head: vT[b][h][dk][s] (16B stores)
//        2 = store f32 [M][N] (+bias) to d_out
template <int ABF16, int MODE>
__global__ __launch_bounds__(256) void gemm_nt(
    const void* __restrict__ Ap, const __bf16* __restrict__ Wbf,
    const float* __restrict__ bias, void* __restrict__ Out) {
  const int lane = threadIdx.x & 31;
  const int wid  = threadIdx.x >> 5;
  const int w    = blockIdx.x * 8 + wid;   // 8192 waves total
  const int m0   = (w >> 4) * 16;          // 512 row tiles
  const int n0   = (w & 15) * 64;          // 16 col strips of 64
  const int lm   = lane & 15;
  const int lh   = lane >> 4;

  const float*  arowf = (const float*)Ap  + (size_t)(m0 + lm) * DD;
  const __bf16* arowb = (const __bf16*)Ap + (size_t)(m0 + lm) * DD;

  v8f acc[4] = {};
  for (int kk = 0; kk < DD; kk += 32) {
    Frag16 a;
    if (ABF16) {
#pragma unroll
      for (int j = 0; j < 8; ++j)
        a.u[j] = *(const unsigned int*)(arowb + kk + kbA(j, lh));
    } else {
#pragma unroll
      for (int j = 0; j < 8; ++j) {
        float2 f = *(const float2*)(arowf + kk + kbA(j, lh));
        a.v[2 * j]     = (__bf16)f.x;
        a.v[2 * j + 1] = (__bf16)f.y;
      }
    }
#pragma unroll
    for (int t = 0; t < 4; ++t) {
      const __bf16* brow = Wbf + (size_t)(n0 + t * 16 + lm) * DD + kk;
      Frag16 bf;
#pragma unroll
      for (int j = 0; j < 8; ++j)
        bf.u[j] = *(const unsigned int*)(brow + kbB(j, lh));
      acc[t] = wmma_bf16(a.v, bf.v, acc[t]);
    }
  }

  // Epilogue. C/D layout: VGPR r holds row m0 + r + 8*lh, column n0+t*16+lm.
#pragma unroll
  for (int t = 0; t < 4; ++t) {
    const int n = n0 + t * 16 + lm;
    const float bn = bias[n];
    if (MODE == 1) {
      // vT[b][h][dk][s]: rows r+8*lh -> 8 consecutive s => one 16B store.
      const int head = n0 >> 6;              // wave-uniform
      const int dk   = t * 16 + lm;
      const int bi   = m0 / SS;              // wave-uniform (tiles don't straddle b)
      const int s0   = (m0 % SS) + 8 * lh;   // multiple of 8 -> 16B aligned
      v8bf pk;
#pragma unroll
      for (int r = 0; r < 8; ++r) pk[r] = (__bf16)(acc[t][r] + bn);
      *(v8bf*)((__bf16*)Out +
               (((size_t)bi * HH + head) * DKK + dk) * SS + s0) = pk;
    } else {
#pragma unroll
      for (int r = 0; r < 8; ++r) {
        const int m = m0 + r + 8 * lh;
        const float val = acc[t][r] + bn;
        if (MODE == 0) {
          ((__bf16*)Out)[(size_t)m * DD + n] = (__bf16)val;
        } else {
          ((float*)Out)[(size_t)m * DD + n] = val;
        }
      }
    }
  }
}

// ---------------- flash attention: one wave per 16-row Q tile ----------------
__global__ __launch_bounds__(256) void attn_kernel(
    const __bf16* __restrict__ qb, const __bf16* __restrict__ kb,
    const __bf16* __restrict__ vtb, const int* __restrict__ mask,
    __bf16* __restrict__ xb) {
  // per-wave LDS slice: 16 rows x 32 bf16 P tile, row stride 34 (pad)
  __shared__ __bf16 plds[8][16 * 34];
  const int lane = threadIdx.x & 31;
  const int wid  = threadIdx.x >> 5;
  const int w    = blockIdx.x * 8 + wid;   // 8192 waves total
  const int qT   = w & (SS / 16 - 1);      // 128 q tiles
  const int bh   = w >> 7;
  const int h    = bh & (HH - 1);
  const int b    = bh >> 4;
  const int lm   = lane & 15;
  const int lh   = lane >> 4;
  __bf16* myLds  = &plds[wid][0];

  // Q fragments (A operand): rows qT*16+lm of this head, K split 0..31 / 32..63
  const __bf16* qrow = qb + (size_t)(b * SS + qT * 16 + lm) * DD + h * DKK;
  Frag16 q0, q1;
#pragma unroll
  for (int j = 0; j < 8; ++j) {
    q0.u[j] = *(const unsigned int*)(qrow + kbA(j, lh));
    q1.u[j] = *(const unsigned int*)(qrow + 32 + kbA(j, lh));
  }

  float rowMax[8], rowSum[8];
#pragma unroll
  for (int r = 0; r < 8; ++r) { rowMax[r] = -__builtin_inff(); rowSum[r] = 0.0f; }
  v8f acc[4] = {};  // 16x64 f32 output accumulator

  const size_t mrowBase = ((size_t)b * SS + qT * 16) * SS;
  const __bf16* vbase   = vtb + (size_t)((b * HH + h) * DKK) * SS;

  for (int j0 = 0; j0 < SS; j0 += 32) {
    // prefetch next kv-block's K rows and V columns (speculative: safe at end)
    __builtin_prefetch(kb + (size_t)(b * SS + j0 + 32 + lm) * DD + h * DKK, 0, 1);
    __builtin_prefetch(vbase + (size_t)lm * SS + j0 + 32, 0, 1);

    // ---- scores: two 16x16 subtiles over DK=64 (2 WMMAs each) ----
    float sc[2][8];
#pragma unroll
    for (int t = 0; t < 2; ++t) {
      const __bf16* krow =
          kb + (size_t)(b * SS + j0 + t * 16 + lm) * DD + h * DKK;
      Frag16 kA, kB;
#pragma unroll
      for (int j = 0; j < 8; ++j) {
        kA.u[j] = *(const unsigned int*)(krow + kbB(j, lh));
        kB.u[j] = *(const unsigned int*)(krow + 32 + kbB(j, lh));
      }
      v8f s = {};
      s = wmma_bf16(q0.v, kA.v, s);
      s = wmma_bf16(q1.v, kB.v, s);
#pragma unroll
      for (int r = 0; r < 8; ++r) {
        const float vv = s[r] * 0.125f;  // 1/sqrt(64)
        const int mk =
            mask[mrowBase + (size_t)(r + 8 * lh) * SS + (j0 + t * 16 + lm)];
        sc[t][r] = (mk == 0) ? NEGV : vv;
      }
    }
    // ---- online softmax over the 32-column block ----
#pragma unroll
    for (int r = 0; r < 8; ++r) {
      float mx = fmaxf(sc[0][r], sc[1][r]);
#pragma unroll
      for (int o = 1; o < 16; o <<= 1) mx = fmaxf(mx, __shfl_xor(mx, o, 32));
      const float nm    = fmaxf(rowMax[r], mx);
      const float alpha = __expf(rowMax[r] - nm);
      rowMax[r] = nm;
      const float p0 = __expf(sc[0][r] - nm);
      const float p1 = __expf(sc[1][r] - nm);
      float ps = p0 + p1;
#pragma unroll
      for (int o = 1; o < 16; o <<= 1) ps += __shfl_xor(ps, o, 32);
      rowSum[r] = rowSum[r] * alpha + ps;
#pragma unroll
      for (int t = 0; t < 4; ++t) acc[t][r] *= alpha;
      // stage P (bf16) into LDS in plain [m][n] layout
      myLds[(r + 8 * lh) * 34 + lm]      = (__bf16)p0;
      myLds[(r + 8 * lh) * 34 + 16 + lm] = (__bf16)p1;
    }
    __builtin_amdgcn_wave_barrier();  // keep LDS write->read ordering
    // ---- repack P as an A fragment (16x32) ----
    Frag16 p;
#pragma unroll
    for (int j = 0; j < 8; ++j)
      p.u[j] = *(const unsigned int*)(myLds + lm * 34 + kbA(j, lh));
    // ---- acc += P @ V  (V^T rows are contiguous in s) ----
#pragma unroll
    for (int t = 0; t < 4; ++t) {
      const __bf16* vrow = vbase + (size_t)(t * 16 + lm) * SS + j0;
      Frag16 vf;
#pragma unroll
      for (int j = 0; j < 8; ++j)
        vf.u[j] = *(const unsigned int*)(vrow + kbB(j, lh));
      acc[t] = wmma_bf16(p.v, vf.v, acc[t]);
    }
    __builtin_amdgcn_wave_barrier();  // WAR: next iter rewrites LDS slice
  }

  // ---- normalize and store x (bf16, [b][s][h*64+dk]) ----
#pragma unroll
  for (int r = 0; r < 8; ++r) {
    const float inv = 1.0f / rowSum[r];
    const size_t orow =
        (size_t)(b * SS + qT * 16 + r + 8 * lh) * DD + h * DKK;
#pragma unroll
    for (int t = 0; t < 4; ++t)
      xb[orow + t * 16 + lm] = (__bf16)(acc[t][r] * inv);
  }
}

// ---------------- host launcher ----------------
extern "C" void kernel_launch(void* const* d_in, const int* in_sizes, int n_in,
                              void* d_out, int out_size, void* d_ws,
                              size_t ws_size, hipStream_t stream) {
  (void)in_sizes; (void)n_in; (void)out_size; (void)ws_size;
  const float* query = (const float*)d_in[0];
  const float* key   = (const float*)d_in[1];
  const float* value = (const float*)d_in[2];
  const int*   mask  = (const int*)d_in[3];
  const float* Wq = (const float*)d_in[4];
  const float* bq = (const float*)d_in[5];
  const float* Wk = (const float*)d_in[6];
  const float* bk = (const float*)d_in[7];
  const float* Wv = (const float*)d_in[8];
  const float* bv = (const float*)d_in[9];
  const float* Wo = (const float*)d_in[10];
  const float* bo = (const float*)d_in[11];

  // workspace layout (bf16): 4 weight mats (8 MB) + q,k,vT,x (64 MB) = 72 MB
  const size_t DDsz = (size_t)DD * DD;
  const size_t ACT  = (size_t)BB * SS * DD;
  __bf16* wqb   = (__bf16*)d_ws;
  __bf16* wkb   = wqb + DDsz;
  __bf16* wvb   = wkb + DDsz;
  __bf16* wob   = wvb + DDsz;
  __bf16* qbuf  = wob + DDsz;
  __bf16* kbuf  = qbuf + ACT;
  __bf16* vtbuf = kbuf + ACT;
  __bf16* xbuf  = vtbuf + ACT;

  cvt_f32_to_bf16<<<512, 256, 0, stream>>>(Wq, wqb, DD * DD);
  cvt_f32_to_bf16<<<512, 256, 0, stream>>>(Wk, wkb, DD * DD);
  cvt_f32_to_bf16<<<512, 256, 0, stream>>>(Wv, wvb, DD * DD);
  cvt_f32_to_bf16<<<512, 256, 0, stream>>>(Wo, wob, DD * DD);

  // projections: 8192 waves each = 1024 blocks x 8 wave32
  gemm_nt<0, 0><<<1024, 256, 0, stream>>>(query, wqb, bq, qbuf);
  gemm_nt<0, 0><<<1024, 256, 0, stream>>>(key,   wkb, bk, kbuf);
  gemm_nt<0, 1><<<1024, 256, 0, stream>>>(value, wvb, bv, vtbuf);

  // attention: B*H*(S/16) = 8192 waves
  attn_kernel<<<1024, 256, 0, stream>>>(qbuf, kbuf, vtbuf, mask, xbuf);

  // output projection -> f32 d_out
  gemm_nt<1, 2><<<1024, 256, 0, stream>>>(xbuf, wob, bo, d_out);
}
